// Perceptron_26938034881178
// MI455X (gfx1250) — matrix operand order repaired and checked by reading, not compiled, not run on hardware
//
#include <hip/hip_runtime.h>

#define B_DIM 512
#define I_DIM 1024
#define O_DIM 256
#define XN (B_DIM * I_DIM)   // 524288 elements of x
#define WN (O_DIM * I_DIM)   // 262144 elements of W

typedef __attribute__((ext_vector_type(16))) __bf16 v16bf;
typedef __attribute__((ext_vector_type(8)))  __bf16 v8bf;
typedef __attribute__((ext_vector_type(8)))  float  v8f;
typedef __attribute__((ext_vector_type(4)))  float  v4f;

// ---------------------------------------------------------------------------
// Pass 1: split f32 -> bf16 hi plane + bf16 residual lo plane, each element
// converted exactly once (instead of 16-32x redundantly inside the GEMM).
// Each thread: 32B f32 in -> 16B hi + 16B lo out.
// ---------------------------------------------------------------------------
__global__ __launch_bounds__(256, 1)
void split_to_bf16_planes(const float* __restrict__ src,
                          __bf16* __restrict__ hiDst,
                          __bf16* __restrict__ loDst)
{
  const int t = blockIdx.x * blockDim.x + threadIdx.x;   // one per 8 elements
  const v4f s0 = *((const v4f*)src + t * 2);
  const v4f s1 = *((const v4f*)src + t * 2 + 1);
  v8bf hi, lo;
#pragma unroll
  for (int e = 0; e < 4; ++e) {
    __bf16 h0 = (__bf16)s0[e];
    hi[e] = h0;
    lo[e] = (__bf16)(s0[e] - (float)h0);
    __bf16 h1 = (__bf16)s1[e];
    hi[4 + e] = h1;
    lo[4 + e] = (__bf16)(s1[e] - (float)h1);
  }
  *(v8bf*)(hiDst + (size_t)t * 8) = hi;
  *(v8bf*)(loDst + (size_t)t * 8) = lo;
}

// ---------------------------------------------------------------------------
// Pass 2: one wave32 per 16x16 output tile of sigmoid(x @ W^T + b).
// Operands come pre-split from the bf16 planes; loop body is pure
// global_load_b128 + v_wmma (3-term split: hi*hi + hi*lo + lo*hi).
// Per-lane operand runs (K-contiguous, 16B aligned):
//   A (m = lane&15, half = lane>>4): runs [half*8 .. +7], [16+half*8 .. +7]
//   B (n = lane&15, half = lane>>4): runs [half*8 .. +7], [16+half*8 .. +7]
// ---------------------------------------------------------------------------
__global__ __launch_bounds__(256, 1)
void perceptron_fwd_wmma(const __bf16* __restrict__ xHi,
                         const __bf16* __restrict__ xLo,
                         const __bf16* __restrict__ wHi,
                         const __bf16* __restrict__ wLo,
                         const float*  __restrict__ bias,
                         float* __restrict__ out)
{
  const int lane = threadIdx.x & 31;
  const int wave = threadIdx.x >> 5;
  const int tile = blockIdx.x * 8 + wave;   // 512 tiles = (512/16)*(256/16)
  const int mBase = (tile >> 4) << 4;
  const int nBase = (tile & 15) << 4;

  const int l15  = lane & 15;
  const int half = lane >> 4;

  const size_t aOff = (size_t)(mBase + l15) * I_DIM + half * 8;
  const size_t bOff = (size_t)(nBase + l15) * I_DIM + half * 8;
  const __bf16* aHiP = xHi + aOff;
  const __bf16* aLoP = xLo + aOff;
  const __bf16* bHiP = wHi + bOff;
  const __bf16* bLoP = wLo + bOff;

  const float bv = bias[nBase + l15];

  v8f acc = {};

#pragma unroll 4
  for (int k = 0; k < I_DIM; k += 32) {
    v8bf ah0 = *(const v8bf*)(aHiP + k);
    v8bf ah1 = *(const v8bf*)(aHiP + k + 16);
    v8bf al0 = *(const v8bf*)(aLoP + k);
    v8bf al1 = *(const v8bf*)(aLoP + k + 16);
    v8bf bh0 = *(const v8bf*)(bHiP + k);
    v8bf bh1 = *(const v8bf*)(bHiP + k + 16);
    v8bf bl0 = *(const v8bf*)(bLoP + k);
    v8bf bl1 = *(const v8bf*)(bLoP + k + 16);

    v16bf aHi = __builtin_shufflevector(ah0, ah1, 0,1,2,3,4,5,6,7,8,9,10,11,12,13,14,15);
    v16bf aLo = __builtin_shufflevector(al0, al1, 0,1,2,3,4,5,6,7,8,9,10,11,12,13,14,15);
    v16bf bHi = __builtin_shufflevector(bh0, bh1, 0,1,2,3,4,5,6,7,8,9,10,11,12,13,14,15);
    v16bf bLo = __builtin_shufflevector(bl0, bl1, 0,1,2,3,4,5,6,7,8,9,10,11,12,13,14,15);

    acc = __builtin_amdgcn_wmma_f32_16x16x32_bf16(false, aHi, false, bHi,
                                                  (short)0, acc, false, false);
    acc = __builtin_amdgcn_wmma_f32_16x16x32_bf16(false, aHi, false, bLo,
                                                  (short)0, acc, false, false);
    acc = __builtin_amdgcn_wmma_f32_16x16x32_bf16(false, aLo, false, bHi,
                                                  (short)0, acc, false, false);
  }

  // C/D layout: VGPR r holds D[r + 8*half][l15]; bias + fast sigmoid.
#pragma unroll
  for (int r = 0; r < 8; ++r) {
    const int row = mBase + r + half * 8;
    const int col = nBase + l15;
    const float v = acc[r] + bv;
    const float e = __expf(-v);
    out[(size_t)row * O_DIM + col] = __builtin_amdgcn_rcpf(1.0f + e);
  }
}

extern "C" void kernel_launch(void* const* d_in, const int* in_sizes, int n_in,
                              void* d_out, int out_size, void* d_ws, size_t ws_size,
                              hipStream_t stream) {
  const float* x    = (const float*)d_in[0];   // (512, 1024) f32
  const float* wgt  = (const float*)d_in[1];   // (256, 1024) f32
  const float* bias = (const float*)d_in[2];   // (256,) f32
  float* out = (float*)d_out;                  // (512, 256) f32

  // Workspace planes (3 MB total): xHi | xLo | wHi | wLo
  __bf16* base = (__bf16*)d_ws;
  __bf16* xHi = base;
  __bf16* xLo = base + XN;
  __bf16* wHi = base + 2 * (size_t)XN;
  __bf16* wLo = base + 2 * (size_t)XN + WN;

  // Pass 1: one-shot f32 -> bf16 hi/lo split (each thread handles 8 floats)
  split_to_bf16_planes<<<XN / 8 / 256, 256, 0, stream>>>(x, xHi, xLo);
  split_to_bf16_planes<<<WN / 8 / 256, 256, 0, stream>>>(wgt, wHi, wLo);

  // Pass 2: WMMA GEMM + bias + sigmoid
  const int tiles  = (B_DIM / 16) * (O_DIM / 16);  // 512
  const int blocks = tiles / 8;                    // 64 blocks x 8 waves
  perceptron_fwd_wmma<<<blocks, 256, 0, stream>>>(xHi, xLo, wHi, wLo, bias, out);
}
